// VectorQuantization_41781441855549
// MI455X (gfx1250) — matrix operand order repaired and checked by reading, not compiled, not run on hardware
//
#include <hip/hip_runtime.h>
#include <hip/hip_bf16.h>

// ---------------------------------------------------------------------------
// VQ nearest-codebook search, CDNA5 (gfx1250) WMMA implementation.
//   score[n,c] = ||e_c||^2 - 2 * z_n . e_c       (||z||^2 dropped: row-const)
//   cross term via bf16 hi/lo split -> 3x v_wmma_f32_16x16x32_bf16 per K-tile
//   B tiles double-buffered in LDS, staged with GLOBAL_LOAD_ASYNC_TO_LDS_B128
//   (ASYNCcnt) when the toolchain exposes it; sync vmem+ds fallback otherwise.
// ---------------------------------------------------------------------------

typedef __attribute__((ext_vector_type(16))) __bf16 v16bf;
typedef __attribute__((ext_vector_type(8)))  __bf16 v8bf;
typedef __attribute__((ext_vector_type(8)))  float  v8f;
typedef __attribute__((ext_vector_type(4)))  int    v4i;

#define LATENT 256
#define NVEC   16384     // 16 * 32 * 32
#define NCODE  8192
#define KSTEPS 8         // 256 / 32
#define CTILES 512       // 8192 / 16

#if defined(__AMDGCN__) &&                                                   \
    __has_builtin(__builtin_amdgcn_global_load_async_to_lds_b128) &&         \
    __has_builtin(__builtin_amdgcn_s_wait_asynccnt)
#define VQ_ASYNC_LDS 1
// Builtin prototype (from hipcc diagnostics): (v4i AS1*, v4i AS3*, Imm, Imm)
#define GLOBAL_AS1(p) ((__attribute__((address_space(1))) v4i*)(p))
#define LDS_AS3(p)    ((__attribute__((address_space(3))) v4i*)(p))
#else
#define VQ_ASYNC_LDS 0
#endif

__device__ __forceinline__ unsigned short f32_to_bf16_rne(float x) {
  unsigned int u = __float_as_uint(x);
  u += 0x7FFFu + ((u >> 16) & 1u);
  return (unsigned short)(u >> 16);
}
__device__ __forceinline__ float bf16_to_f32(unsigned short h) {
  return __uint_as_float(((unsigned int)h) << 16);
}

// ---------------------------------------------------------------------------
// Prep 1: codebook -> scaled (-2e) hi/lo bf16 + exact fp32 norms ||e||^2.
// ---------------------------------------------------------------------------
__global__ void __launch_bounds__(256) vq_prep_codebook(
    const float* __restrict__ cb,
    unsigned short* __restrict__ cb_hi,
    unsigned short* __restrict__ cb_lo,
    float* __restrict__ norms) {
  const int c = blockIdx.x;
  const int t = threadIdx.x;
  const size_t o = (size_t)c * LATENT + t;
  float v = cb[o];
  float s = -2.0f * v;
  unsigned short hi = f32_to_bf16_rne(s);
  float lo = s - bf16_to_f32(hi);
  cb_hi[o] = hi;
  cb_lo[o] = f32_to_bf16_rne(lo);

  __shared__ float red[256];
  red[t] = v * v;
  __syncthreads();
  for (int off = 128; off > 0; off >>= 1) {
    if (t < off) red[t] += red[t + off];
    __syncthreads();
  }
  if (t == 0) norms[c] = red[0];
}

// ---------------------------------------------------------------------------
// Prep 2: z_e NCHW (16,256,32,32) -> zt (N=16384, 256) in hi/lo bf16.
// ---------------------------------------------------------------------------
__global__ void __launch_bounds__(1024) vq_prep_z(
    const float* __restrict__ z,
    unsigned short* __restrict__ zt_hi,
    unsigned short* __restrict__ zt_lo) {
  __shared__ float tile[32][33];
  const int b  = blockIdx.z;
  const int d0 = blockIdx.y * 32;
  const int p0 = blockIdx.x * 32;
  const int tx = threadIdx.x, ty = threadIdx.y;

  tile[ty][tx] = z[((size_t)(b * 256 + d0 + ty)) * 1024 + p0 + tx];
  __syncthreads();
  float v = tile[tx][ty];
  unsigned short hi = f32_to_bf16_rne(v);
  float lo = v - bf16_to_f32(hi);
  const size_t o = ((size_t)(b * 1024 + p0 + ty)) * LATENT + d0 + tx;
  zt_hi[o] = hi;
  zt_lo[o] = f32_to_bf16_rne(lo);
}

// ---------------------------------------------------------------------------
// B-tile staging: 16 codebook rows (hi+lo) = 2 x 8 KB -> LDS.
// Async path: 4 x global_load_async_to_lds_b128 per thread (ASYNCcnt).
// Fallback:   uint4 loads + ds stores.
// ---------------------------------------------------------------------------
__device__ __forceinline__ void vq_stage_tile(
    const unsigned short* __restrict__ gh,
    const unsigned short* __restrict__ gl,
    unsigned short* lh, unsigned short* ll, int tid) {
#if VQ_ASYNC_LDS
#pragma unroll
  for (int j = 0; j < 2; ++j) {
    const int e = (tid + j * 256) * 8;   // 16-byte granularity (8 ushorts)
    __builtin_amdgcn_global_load_async_to_lds_b128(
        GLOBAL_AS1(gh + e), LDS_AS3(lh + e), 0, 0);
    __builtin_amdgcn_global_load_async_to_lds_b128(
        GLOBAL_AS1(gl + e), LDS_AS3(ll + e), 0, 0);
  }
#else
  const uint4* sh = (const uint4*)gh;
  const uint4* sl = (const uint4*)gl;
  uint4* dh = (uint4*)lh;
  uint4* dl = (uint4*)ll;
#pragma unroll
  for (int j = 0; j < 2; ++j) {
    dh[tid + j * 256] = sh[tid + j * 256];
    dl[tid + j * 256] = sl[tid + j * 256];
  }
#endif
}

__device__ __forceinline__ void vq_wait_async() {
#if VQ_ASYNC_LDS
  __builtin_amdgcn_s_wait_asynccnt(0);
#endif
}

// ---------------------------------------------------------------------------
// Main: fused GEMM + argmin + gather.
// 256 threads = 8 waves; wave owns 16 rows (WMMA M); WG owns 128 rows.
// A (16x256 hi+lo) hoisted to VGPRs; B tile (16 codes x 256, hi+lo) staged
// through double-buffered LDS, shared by all 8 waves.
// ---------------------------------------------------------------------------
__global__ void __launch_bounds__(256) vq_main(
    const unsigned short* __restrict__ zt_hi,
    const unsigned short* __restrict__ zt_lo,
    const unsigned short* __restrict__ cb_hi,
    const unsigned short* __restrict__ cb_lo,
    const float* __restrict__ norms,
    const float* __restrict__ cb_f32,
    float* __restrict__ out_zq,
    float* __restrict__ out_idx) {
  __shared__ __align__(16) unsigned short lbh[2][16 * LATENT];  // 2 x 8 KB
  __shared__ __align__(16) unsigned short lbl[2][16 * LATENT];  // 2 x 8 KB

  const int tid     = threadIdx.x;
  const int lane    = tid & 31;
  const int wave    = tid >> 5;
  const int rowbase = blockIdx.x * 128 + wave * 16;
  const int m       = lane & 15;       // A: row within tile / B,C: column N
  const int khalf   = lane >> 4;       // 0 or 1 (K half selector)

  // --- Hoist A fragments (z rows, hi+lo) into VGPRs: 16 x v16bf = 128 VGPRs.
  // A k order used here (contiguous, same as B): lanes 0-15 hold K{0..15}, lanes 16-31 K{16..31}.
  v16bf a_hi[KSTEPS], a_lo[KSTEPS];
  {
    const unsigned short* rh = zt_hi + (size_t)(rowbase + m) * LATENT;
    const unsigned short* rl = zt_lo + (size_t)(rowbase + m) * LATENT;
    const int ksel = khalf * 16;
#pragma unroll
    for (int k = 0; k < KSTEPS; ++k) {
      v8bf h0 = *(const v8bf*)(rh + k * 32 + ksel);
      v8bf h1 = *(const v8bf*)(rh + k * 32 + 8 + ksel);
      a_hi[k] = __builtin_shufflevector(h0, h1, 0,1,2,3,4,5,6,7,8,9,10,11,12,13,14,15);
      v8bf l0 = *(const v8bf*)(rl + k * 32 + ksel);
      v8bf l1 = *(const v8bf*)(rl + k * 32 + 8 + ksel);
      a_lo[k] = __builtin_shufflevector(l0, l1, 0,1,2,3,4,5,6,7,8,9,10,11,12,13,14,15);
    }
  }

  float bestv[8];
  int   besti[8];
#pragma unroll
  for (int r = 0; r < 8; ++r) { bestv[r] = 3.4e38f; besti[r] = 0; }

  // Prime the pipeline: stage tile 0 into buffer 0.
  vq_stage_tile(cb_hi, cb_lo, lbh[0], lbl[0], tid);

  for (int ct = 0; ct < CTILES; ++ct) {
    const int cur = ct & 1;
    vq_wait_async();     // this wave's async writes for tile ct have landed
    __syncthreads();     // all waves' writes visible; buffer cur^1 fully consumed

    if (ct + 1 < CTILES) {
      const size_t off = (size_t)(ct + 1) * 16 * LATENT;
      vq_stage_tile(cb_hi + off, cb_lo + off, lbh[cur ^ 1], lbl[cur ^ 1], tid);
    }

    const unsigned short* bh_base = lbh[cur];
    const unsigned short* bl_base = lbl[cur];
    v8f acc = {0.f, 0.f, 0.f, 0.f, 0.f, 0.f, 0.f, 0.f};
    // B k order used here (contiguous, same as A): lanes 0-15 K=0..15 (N=lane), lanes 16-31 K=16..31.
#pragma unroll
    for (int k = 0; k < KSTEPS; ++k) {
      v16bf bh = *(const v16bf*)(bh_base + m * LATENT + k * 32 + khalf * 16);
      v16bf bl = *(const v16bf*)(bl_base + m * LATENT + k * 32 + khalf * 16);
      acc = __builtin_amdgcn_wmma_f32_16x16x32_bf16(false, a_hi[k], false, bh,
                                                    (short)0, acc, false, false);
      acc = __builtin_amdgcn_wmma_f32_16x16x32_bf16(false, a_hi[k], false, bl,
                                                    (short)0, acc, false, false);
      acc = __builtin_amdgcn_wmma_f32_16x16x32_bf16(false, a_lo[k], false, bh,
                                                    (short)0, acc, false, false);
    }

    const int   c   = ct * 16 + m;
    const float nrm = norms[c];
#pragma unroll
    for (int r = 0; r < 8; ++r) {
      float d = acc[r] + nrm;     // ||e||^2 - 2 z.e
      if (d < bestv[r]) { bestv[r] = d; besti[r] = c; }
    }
  }

  // Cross-lane argmin: butterfly within each 16-lane half (rows 0-7 / 8-15).
#pragma unroll
  for (int dmask = 1; dmask < 16; dmask <<= 1) {
#pragma unroll
    for (int r = 0; r < 8; ++r) {
      float ov = __shfl_xor(bestv[r], dmask, 32);
      int   oi = __shfl_xor(besti[r], dmask, 32);
      if (ov < bestv[r] || (ov == bestv[r] && oi < besti[r])) {
        bestv[r] = ov; besti[r] = oi;
      }
    }
  }

  // Gather winning fp32 codebook rows + write indices (as fp32 values).
#pragma unroll
  for (int rr = 0; rr < 16; ++rr) {
    const int idx = __shfl(besti[rr & 7], (rr < 8) ? 0 : 16, 32);
    const float4* src = (const float4*)(cb_f32 + (size_t)idx * LATENT);
    float4* dst = (float4*)(out_zq + (size_t)(rowbase + rr) * LATENT);
    dst[lane]      = src[lane];
    dst[lane + 32] = src[lane + 32];
    if (lane == 0) out_idx[rowbase + rr] = (float)idx;
  }
}

// ---------------------------------------------------------------------------
extern "C" void kernel_launch(void* const* d_in, const int* in_sizes, int n_in,
                              void* d_out, int out_size, void* d_ws, size_t ws_size,
                              hipStream_t stream) {
  const float* z_e      = (const float*)d_in[0];   // (16,256,32,32) f32
  const float* codebook = (const float*)d_in[1];   // (8192,256) f32

  char* ws = (char*)d_ws;
  unsigned short* cb_hi = (unsigned short*)(ws);                         // 4 MB
  unsigned short* cb_lo = (unsigned short*)(ws + (size_t)(4 << 20));     // 4 MB
  float*          norms = (float*)(ws + (size_t)(8 << 20));              // 32 KB
  unsigned short* zt_hi = (unsigned short*)(ws + (size_t)(8 << 20) + (64 << 10)); // 8 MB
  unsigned short* zt_lo = (unsigned short*)(ws + (size_t)(16 << 20) + (64 << 10)); // 8 MB

  float* out_zq  = (float*)d_out;
  float* out_idx = out_zq + (size_t)NVEC * LATENT;

  vq_prep_codebook<<<NCODE, 256, 0, stream>>>(codebook, cb_hi, cb_lo, norms);
  vq_prep_z<<<dim3(32, 8, 16), dim3(32, 32), 0, stream>>>(z_e, zt_hi, zt_lo);
  vq_main<<<NVEC / 128, 256, 0, stream>>>(zt_hi, zt_lo, cb_hi, cb_lo, norms,
                                          codebook, out_zq, out_idx);
}
